// VariationalAttention_53352083750969
// MI455X (gfx1250) — matrix operand order repaired and checked by reading, not compiled, not run on hardware
//
#include <hip/hip_runtime.h>
#include <hip/hip_bf16.h>
#include <math.h>

// Problem dims (fixed by the reference)
#define BDIM 16
#define TDIM 1024
#define SDIM 2048
#define DDIM 512

typedef __bf16 bf16_t;
typedef __attribute__((ext_vector_type(16))) __bf16 v16bf;
typedef __attribute__((ext_vector_type(8)))  __bf16 v8bf;
typedef __attribute__((ext_vector_type(8)))  float  v8f;

constexpr int BM = 128, BN = 128, BK = 32;
constexpr int LDT = 40;        // padded bf16 tile row stride: 80 bytes (16B multiple)
constexpr int NTHREADS = 256;  // 8 waves (wave32)

// ---------------------------------------------------------------------------
// LDS helpers.  Tiles stored "outer-major x K": tile[r][k], r<128, k<32,
// row stride LDT.  A-operand: r == M row.  B-operand: r == N column (staged
// K-transposed) so each WMMA fragment is two contiguous 16B LDS reads/lane.
// ---------------------------------------------------------------------------
__device__ __forceinline__ uint32_t lds_off(const void* p) {
  // For LDS apertures the low 32 bits of the flat address are the
  // wave-relative LDS byte offset (CDNA5 ISA 10.2 aperture mapping).
  return (uint32_t)(uintptr_t)p;
}

// Async global->LDS copy of one 128x32 bf16 tile (2 b128 instrs per wave).
__device__ __forceinline__ void async_tile_bf16(uint32_t ldsTile,
                                                const bf16_t* src, int ld,
                                                int tid) {
#pragma unroll
  for (int it = 0; it < 2; ++it) {
    int idx = tid + it * NTHREADS;       // 0..511 chunks of 8 bf16 (16B)
    int r   = idx >> 2;                  // 4 chunks per 32-elem row
    int c   = (idx & 3) << 3;
    uint32_t loff = ldsTile + (uint32_t)((r * LDT + c) * 2);
    const bf16_t* g = src + (size_t)r * ld + c;
    asm volatile("global_load_async_to_lds_b128 %0, %1, off"
                 :: "v"(loff), "v"(g) : "memory");
  }
}

__device__ __forceinline__ void wait_async_4() {
  asm volatile("s_wait_asynccnt 0x4" ::: "memory");
}
__device__ __forceinline__ void wait_async_2() {
  asm volatile("s_wait_asynccnt 0x2" ::: "memory");
}
__device__ __forceinline__ void wait_async_0() {
  asm volatile("s_wait_asynccnt 0x0" ::: "memory");
}

// Synchronous f32 -> bf16 staging (only used for the softmax output operand).
__device__ __forceinline__ void stage_rm_f32(bf16_t* tile, const float* src,
                                             int ld, int tid) {
#pragma unroll
  for (int it = 0; it < 4; ++it) {
    int idx = tid + it * NTHREADS;       // 0..1023 float4 units
    int r   = idx >> 3;
    int c   = (idx & 7) << 2;
    float4 v = *reinterpret_cast<const float4*>(src + (size_t)r * ld + c);
    bf16_t* d = tile + r * LDT + c;
    d[0] = (bf16_t)v.x; d[1] = (bf16_t)v.y; d[2] = (bf16_t)v.z; d[3] = (bf16_t)v.w;
  }
}

// ---------------------------------------------------------------------------
// WMMA fragment loads (CDNA5 16x16x32 bf16 lane layouts, wave32).
// ---------------------------------------------------------------------------
__device__ __forceinline__ v16bf load_fragA(const bf16_t* tile, int mBase, int lane) {
  int half = lane >> 4, m = lane & 15;
  const bf16_t* row = tile + (mBase + m) * LDT;
  v8bf c0 = *reinterpret_cast<const v8bf*>(row + half * 8);
  v8bf c1 = *reinterpret_cast<const v8bf*>(row + 16 + half * 8);
  v16bf f;
#pragma unroll
  for (int i = 0; i < 8; ++i) { f[i] = c0[i]; f[8 + i] = c1[i]; }
  return f;
}

__device__ __forceinline__ v16bf load_fragB(const bf16_t* tile, int nBase, int lane) {
  int half = lane >> 4, n = lane & 15;
  const bf16_t* row = tile + (nBase + n) * LDT;
  v8bf c0 = *reinterpret_cast<const v8bf*>(row + half * 8);
  v8bf c1 = *reinterpret_cast<const v8bf*>(row + 16 + half * 8);
  v16bf f;
#pragma unroll
  for (int i = 0; i < 8; ++i) { f[i] = c0[i]; f[8 + i] = c1[i]; }
  return f;
}

// One BK=32 step: each wave computes a 32x64 patch = 2x4 WMMA tiles.
__device__ __forceinline__ void mma_step(const bf16_t* At, const bf16_t* Bt,
                                         int lane, int wm, int wn, v8f acc[2][4]) {
  v16bf a[2], bb[4];
#pragma unroll
  for (int i = 0; i < 2; ++i) a[i] = load_fragA(At, wm * 32 + i * 16, lane);
#pragma unroll
  for (int j = 0; j < 4; ++j) bb[j] = load_fragB(Bt, wn * 64 + j * 16, lane);
#pragma unroll
  for (int i = 0; i < 2; ++i)
#pragma unroll
    for (int j = 0; j < 4; ++j)
      acc[i][j] = __builtin_amdgcn_wmma_f32_16x16x32_bf16(
          false, a[i], false, bb[j], (short)0, acc[i][j], false, false);
}

// Double-buffered fully-async bf16 GEMM inner loop.
__device__ __forceinline__ void gemm_loop_async(const bf16_t* Asrc, int lda,
                                                const bf16_t* Bsrc, int ldb,
                                                int ktot, int tid, int lane,
                                                int wm, int wn, v8f acc[2][4],
                                                bf16_t At[2][BM * LDT],
                                                bf16_t Bt[2][BM * LDT]) {
  uint32_t aOff[2] = { lds_off(At[0]), lds_off(At[1]) };
  uint32_t bOff[2] = { lds_off(Bt[0]), lds_off(Bt[1]) };
  async_tile_bf16(aOff[0], Asrc, lda, tid);
  async_tile_bf16(bOff[0], Bsrc, ldb, tid);
  int nsteps = ktot / BK;
  for (int s = 0; s < nsteps; ++s) {
    int cur = s & 1, nxt = cur ^ 1;
    if (s + 1 < nsteps) {
      async_tile_bf16(aOff[nxt], Asrc + (size_t)(s + 1) * BK, lda, tid);
      async_tile_bf16(bOff[nxt], Bsrc + (size_t)(s + 1) * BK, ldb, tid);
      wait_async_4();                 // stage s landed; stage s+1 in flight
    } else {
      wait_async_0();
    }
    __syncthreads();
    mma_step(At[cur], Bt[cur], lane, wm, wn, acc);
    __syncthreads();                  // buffer `cur` free for reuse
  }
}

// ---------------------------------------------------------------------------
// Conversion / layout kernels (one-time, ~5us of HBM traffic total)
// ---------------------------------------------------------------------------
__global__ __launch_bounds__(NTHREADS)
void k_cvt(const float* __restrict__ src, bf16_t* __restrict__ dst, int n) {
  int i = blockIdx.x * NTHREADS + threadIdx.x;
  if (i < n) dst[i] = (bf16_t)src[i];
}

// bf16 input into the high half of the concat buffer [B,T,2D]
__global__ __launch_bounds__(NTHREADS)
void k_fill_concat(const float* __restrict__ inp, bf16_t* __restrict__ cat) {
  size_t i = (size_t)blockIdx.x * NTHREADS + threadIdx.x;   // over B*T*D
  size_t row = i / DDIM;
  int d = (int)(i % DDIM);
  cat[row * (2 * DDIM) + DDIM + d] = (bf16_t)inp[i];
}

// memT[b,d,s] = (bf16) mem[b,s,d]   (LDS-tiled 32x32 transpose)
__global__ __launch_bounds__(NTHREADS)
void k_transpose(const float* __restrict__ mem, bf16_t* __restrict__ memT) {
  __shared__ float tile[32][33];
  int b = blockIdx.z;
  int s0 = blockIdx.x * 32, d0 = blockIdx.y * 32;
  int tx = threadIdx.x & 31, ty = threadIdx.x >> 5;   // 32 x 8
  const float* src = mem + ((size_t)b * SDIM + s0) * DDIM + d0;
#pragma unroll
  for (int i = 0; i < 32; i += 8)
    tile[ty + i][tx] = src[(size_t)(ty + i) * DDIM + tx];
  __syncthreads();
  bf16_t* dst = memT + ((size_t)b * DDIM + d0) * SDIM + s0;
#pragma unroll
  for (int i = 0; i < 32; i += 8)
    dst[(size_t)(ty + i) * SDIM + tx] = (bf16_t)tile[tx][ty + i];
}

// ---------------------------------------------------------------------------
// Kernel 1: h_t[b,t,e] = sum_d input[b,t,d]*W_in[e,d]  (bf16 -> ws)
// A from concat high half (bf16 input), B = W_in bf16.
// ---------------------------------------------------------------------------
__global__ __launch_bounds__(NTHREADS)
void k_gemm_ht(const bf16_t* __restrict__ cat, const bf16_t* __restrict__ winb,
               bf16_t* __restrict__ ht) {
  __shared__ __align__(16) bf16_t At[2][BM * LDT];
  __shared__ __align__(16) bf16_t Bt[2][BM * LDT];
  int tid = threadIdx.x, lane = tid & 31, wave = tid >> 5;
  int wm = wave & 3, wn = wave >> 2;
  int b = blockIdx.z, mBase = blockIdx.y * BM, nBase = blockIdx.x * BN;
  const bf16_t* Asrc = cat + (size_t)(b * TDIM + mBase) * (2 * DDIM) + DDIM;
  const bf16_t* Bsrc = winb + (size_t)nBase * DDIM;   // B[k][n] = W_in[n*D+k]
  v8f acc[2][4] = {};
  gemm_loop_async(Asrc, 2 * DDIM, Bsrc, DDIM, DDIM, tid, lane, wm, wn, acc, At, Bt);
  int n = lane & 15, half = lane >> 4;
  bf16_t* o = ht + (size_t)(b * TDIM + mBase) * DDIM + nBase;
#pragma unroll
  for (int i = 0; i < 2; ++i)
#pragma unroll
    for (int j = 0; j < 4; ++j) {
      int col = wn * 64 + j * 16 + n;
#pragma unroll
      for (int r = 0; r < 8; ++r) {
        int row = wm * 32 + i * 16 + half * 8 + r;
        o[(size_t)row * DDIM + col] = (bf16_t)acc[i][j][r];
      }
    }
}

// ---------------------------------------------------------------------------
// Kernel 2: scores[b,t,s] = sum_d h_t[b,t,d]*mem[b,s,d]; masked -> align buf
// ---------------------------------------------------------------------------
__global__ __launch_bounds__(NTHREADS)
void k_gemm_scores(const bf16_t* __restrict__ ht, const bf16_t* __restrict__ memb,
                   const int* __restrict__ lens, float* __restrict__ alignBuf) {
  __shared__ __align__(16) bf16_t At[2][BM * LDT];
  __shared__ __align__(16) bf16_t Bt[2][BM * LDT];
  int tid = threadIdx.x, lane = tid & 31, wave = tid >> 5;
  int wm = wave & 3, wn = wave >> 2;
  int b = blockIdx.z, mBase = blockIdx.y * BM, nBase = blockIdx.x * BN;
  const bf16_t* Asrc = ht + (size_t)(b * TDIM + mBase) * DDIM;
  const bf16_t* Bsrc = memb + (size_t)(b * SDIM + nBase) * DDIM; // B[k][n]=mem[n][k]
  v8f acc[2][4] = {};
  gemm_loop_async(Asrc, DDIM, Bsrc, DDIM, DDIM, tid, lane, wm, wn, acc, At, Bt);
  int len = lens[b];
  int n = lane & 15, half = lane >> 4;
  float* o = alignBuf + (size_t)(b * TDIM + mBase) * SDIM + nBase;
#pragma unroll
  for (int i = 0; i < 2; ++i)
#pragma unroll
    for (int j = 0; j < 4; ++j) {
      int col = wn * 64 + j * 16 + n;
      bool in = (nBase + col) < len;
#pragma unroll
      for (int r = 0; r < 8; ++r) {
        int row = wm * 32 + i * 16 + half * 8 + r;
        o[(size_t)row * SDIM + col] = in ? acc[i][j][r] : -1.0e30f;
      }
    }
}

// ---------------------------------------------------------------------------
// Kernel 3: in-place row softmax over S=2048 (one block per (b,t) row)
// ---------------------------------------------------------------------------
__global__ __launch_bounds__(NTHREADS)
void k_softmax(float* __restrict__ buf) {
  __shared__ float red[NTHREADS];
  int row = blockIdx.x, tid = threadIdx.x;
  float* p = buf + (size_t)row * SDIM;
  float x[SDIM / NTHREADS];
  float m = -3.0e38f;
#pragma unroll
  for (int i = 0; i < SDIM / NTHREADS; ++i) {
    x[i] = p[tid + i * NTHREADS];
    m = fmaxf(m, x[i]);
  }
  red[tid] = m; __syncthreads();
  for (int off = NTHREADS / 2; off > 0; off >>= 1) {
    if (tid < off) red[tid] = fmaxf(red[tid], red[tid + off]);
    __syncthreads();
  }
  m = red[0]; __syncthreads();
  float sum = 0.f;
#pragma unroll
  for (int i = 0; i < SDIM / NTHREADS; ++i) {
    x[i] = __expf(x[i] - m);
    sum += x[i];
  }
  red[tid] = sum; __syncthreads();
  for (int off = NTHREADS / 2; off > 0; off >>= 1) {
    if (tid < off) red[tid] += red[tid + off];
    __syncthreads();
  }
  float inv = 1.0f / red[0];
#pragma unroll
  for (int i = 0; i < SDIM / NTHREADS; ++i) p[tid + i * NTHREADS] = x[i] * inv;
}

// ---------------------------------------------------------------------------
// Kernel 4: context[b,t,d] = sum_s align[b,t,s]*mem[b,s,d] -> bf16 into the
// low half of the concat buffer.  A staged sync (f32 source), B async from
// the transposed bf16 memory bank memT[b,d,s].
// ---------------------------------------------------------------------------
__global__ __launch_bounds__(NTHREADS)
void k_gemm_ctx(const float* __restrict__ alignBuf, const bf16_t* __restrict__ memT,
                bf16_t* __restrict__ cat) {
  __shared__ __align__(16) bf16_t At[2][BM * LDT];
  __shared__ __align__(16) bf16_t Bt[2][BM * LDT];
  int tid = threadIdx.x, lane = tid & 31, wave = tid >> 5;
  int wm = wave & 3, wn = wave >> 2;
  int b = blockIdx.z, mBase = blockIdx.y * BM, nBase = blockIdx.x * BN;
  const float*  Asrc = alignBuf + (size_t)(b * TDIM + mBase) * SDIM;
  const bf16_t* Bsrc = memT + (size_t)(b * DDIM + nBase) * SDIM; // B[k][n]=memT[n][k]
  uint32_t bOff[2] = { lds_off(Bt[0]), lds_off(Bt[1]) };
  async_tile_bf16(bOff[0], Bsrc, SDIM, tid);
  v8f acc[2][4] = {};
  constexpr int NSTEPS = SDIM / BK;
  for (int s = 0; s < NSTEPS; ++s) {
    int cur = s & 1, nxt = cur ^ 1;
    if (s + 1 < NSTEPS) async_tile_bf16(bOff[nxt], Bsrc + (size_t)(s + 1) * BK, SDIM, tid);
    stage_rm_f32(At[cur], Asrc + (size_t)s * BK, SDIM, tid);
    if (s + 1 < NSTEPS) wait_async_2(); else wait_async_0();
    __syncthreads();
    mma_step(At[cur], Bt[cur], lane, wm, wn, acc);
    __syncthreads();
  }
  int n = lane & 15, half = lane >> 4;
  bf16_t* o = cat + (size_t)(b * TDIM + mBase) * (2 * DDIM) + nBase;
#pragma unroll
  for (int i = 0; i < 2; ++i)
#pragma unroll
    for (int j = 0; j < 4; ++j) {
      int col = wn * 64 + j * 16 + n;
#pragma unroll
      for (int r = 0; r < 8; ++r) {
        int row = wm * 32 + i * 16 + half * 8 + r;
        o[(size_t)row * (2 * DDIM) + col] = (bf16_t)acc[i][j][r];
      }
    }
}

// ---------------------------------------------------------------------------
// Kernel 5: h_c[b,t,d] = tanh( sum_k concat[b,t,k]*W_out[d,k] ), K = 2D
// ---------------------------------------------------------------------------
__global__ __launch_bounds__(NTHREADS)
void k_gemm_hc(const bf16_t* __restrict__ cat, const bf16_t* __restrict__ woutb,
               float* __restrict__ hc) {
  __shared__ __align__(16) bf16_t At[2][BM * LDT];
  __shared__ __align__(16) bf16_t Bt[2][BM * LDT];
  int tid = threadIdx.x, lane = tid & 31, wave = tid >> 5;
  int wm = wave & 3, wn = wave >> 2;
  int b = blockIdx.z, mBase = blockIdx.y * BM, nBase = blockIdx.x * BN;
  const bf16_t* Asrc = cat + (size_t)(b * TDIM + mBase) * (2 * DDIM);
  const bf16_t* Bsrc = woutb + (size_t)nBase * (2 * DDIM); // B[k][n]=W_out[n*2D+k]
  v8f acc[2][4] = {};
  gemm_loop_async(Asrc, 2 * DDIM, Bsrc, 2 * DDIM, 2 * DDIM, tid, lane, wm, wn, acc, At, Bt);
  int n = lane & 15, half = lane >> 4;
  float* o = hc + (size_t)(b * TDIM + mBase) * DDIM + nBase;
#pragma unroll
  for (int i = 0; i < 2; ++i)
#pragma unroll
    for (int j = 0; j < 4; ++j) {
      int col = wn * 64 + j * 16 + n;
#pragma unroll
      for (int r = 0; r < 8; ++r) {
        int row = wm * 32 + i * 16 + half * 8 + r;
        o[(size_t)row * DDIM + col] = tanhf(acc[i][j][r]);
      }
    }
}

// ---------------------------------------------------------------------------
// Kernel 6: Gumbel-argmax categorical sample per (b,t) row (K=1, seed 42)
// ---------------------------------------------------------------------------
__global__ __launch_bounds__(NTHREADS)
void k_sample(const float* __restrict__ qla, float* __restrict__ qs,
              float* __restrict__ slp) {
  __shared__ float sv[NTHREADS];
  __shared__ int   si[NTHREADS];
  int row = blockIdx.x, tid = threadIdx.x;
  const float* lp = qla + (size_t)row * SDIM;
  float best = -3.0e38f; int bidx = 0;
#pragma unroll
  for (int i = 0; i < SDIM / NTHREADS; ++i) {
    int s = tid + i * NTHREADS;
    float l = lp[s];
    unsigned h = (unsigned)row * 2654435761u + (unsigned)s;
    h ^= 0x9E3779B9u + 42u;
    h *= 0x85EBCA6Bu; h ^= h >> 13;
    h *= 0xC2B2AE35u; h ^= h >> 16;
    float u = ((float)(h >> 8) + 0.5f) * (1.0f / 16777216.0f); // (0,1)
    float g = -__logf(-__logf(u));
    float v = l + g;
    if (v > best) { best = v; bidx = s; }
  }
  sv[tid] = best; si[tid] = bidx;
  __syncthreads();
  for (int off = NTHREADS / 2; off > 0; off >>= 1) {
    if (tid < off && sv[tid + off] > sv[tid]) {
      sv[tid] = sv[tid + off]; si[tid] = si[tid + off];
    }
    __syncthreads();
  }
  int sel = si[0];
  float* qrow = qs + (size_t)row * SDIM;
#pragma unroll
  for (int i = 0; i < SDIM / NTHREADS; ++i) {
    int s = tid + i * NTHREADS;
    qrow[s] = (s == sel) ? 1.0f : 0.0f;
  }
  if (tid == 0) slp[row] = lp[sel];
}

// ---------------------------------------------------------------------------
extern "C" void kernel_launch(void* const* d_in, const int* in_sizes, int n_in,
                              void* d_out, int out_size, void* d_ws, size_t ws_size,
                              hipStream_t stream) {
  (void)in_sizes; (void)n_in; (void)out_size; (void)ws_size;
  const float* inp  = (const float*)d_in[0];   // [B,T,D]
  const float* mem  = (const float*)d_in[1];   // [B,S,D]
  const int*   lens = (const int*)d_in[2];     // [B]
  const float* qla  = (const float*)d_in[4];   // q_log_alpha [B,T,S]
  const float* Win  = (const float*)d_in[5];   // [D,D]
  const float* Wout = (const float*)d_in[6];   // [D,2D]

  float* out      = (float*)d_out;
  float* hcOut    = out;                                     // B*T*D
  float* alignOut = hcOut + (size_t)BDIM * TDIM * DDIM;      // B*T*S
  float* qsOut    = alignOut + (size_t)BDIM * TDIM * SDIM;   // B*T*S
  float* slpOut   = qsOut + (size_t)BDIM * TDIM * SDIM;      // B*T

  bf16_t* ht    = (bf16_t*)d_ws;                             // B*T*D
  bf16_t* cat   = ht   + (size_t)BDIM * TDIM * DDIM;         // B*T*2D
  bf16_t* memb  = cat  + (size_t)BDIM * TDIM * 2 * DDIM;     // B*S*D
  bf16_t* memT  = memb + (size_t)BDIM * SDIM * DDIM;         // B*D*S
  bf16_t* winb  = memT + (size_t)BDIM * SDIM * DDIM;         // D*D
  bf16_t* woutb = winb + (size_t)DDIM * DDIM;                // D*2D

  dim3 blk(NTHREADS);
  // one-time layout/precision conversion
  k_cvt        <<<(BDIM * SDIM * DDIM) / NTHREADS, blk, 0, stream>>>(mem, memb, BDIM * SDIM * DDIM);
  k_transpose  <<<dim3(SDIM / 32, DDIM / 32, BDIM), blk, 0, stream>>>(mem, memT);
  k_cvt        <<<(DDIM * DDIM) / NTHREADS, blk, 0, stream>>>(Win, winb, DDIM * DDIM);
  k_cvt        <<<(DDIM * 2 * DDIM) / NTHREADS, blk, 0, stream>>>(Wout, woutb, DDIM * 2 * DDIM);
  k_fill_concat<<<(BDIM * TDIM * DDIM) / NTHREADS, blk, 0, stream>>>(inp, cat);
  // attention pipeline
  k_gemm_ht    <<<dim3(DDIM / BN, TDIM / BM, BDIM), blk, 0, stream>>>(cat, winb, ht);
  k_gemm_scores<<<dim3(SDIM / BN, TDIM / BM, BDIM), blk, 0, stream>>>(ht, memb, lens, alignOut);
  k_softmax    <<<BDIM * TDIM, blk, 0, stream>>>(alignOut);
  k_gemm_ctx   <<<dim3(DDIM / BN, TDIM / BM, BDIM), blk, 0, stream>>>(alignOut, memT, cat);
  k_gemm_hc    <<<dim3(DDIM / BN, TDIM / BM, BDIM), blk, 0, stream>>>(cat, woutb, hcOut);
  k_sample     <<<BDIM * TDIM, blk, 0, stream>>>(qla, qsOut, slpOut);
}
